// MambaBackbone_17755394801698
// MI455X (gfx1250) — matrix operand order repaired and checked by reading, not compiled, not run on hardware
//
#include <hip/hip_runtime.h>
#include <hip/hip_bf16.h>
#include <stdint.h>

// ---------------- problem constants (from reference) ----------------
#define DMODEL   512
#define DINNER   1024
#define DSTATE   16
#define DCONV    4
#define DTRANK   32
#define BATCH    4
#define LSEQ     1024
#define NLAYERS  2
#define XPROJ_N  (DTRANK + 2*DSTATE)   // 64
#define MROWS    (BATCH*LSEQ)          // 4096

typedef __bf16 bf16_t;
typedef __attribute__((ext_vector_type(16))) __bf16 v16bf;
typedef __attribute__((ext_vector_type(8)))  float  v8f;

// ---------------- helpers ----------------
struct Frag16 { uint32_t u[8]; };

__device__ __forceinline__ v16bf frag_as_v16bf(const Frag16 &f) {
    union { Frag16 f; v16bf v; } c; c.f = f; return c.v;
}
__device__ __forceinline__ void load8bf(uint32_t *dst, const bf16_t *p) {
    const uint4 t = *reinterpret_cast<const uint4 *>(p);   // global_load_b128
    dst[0] = t.x; dst[1] = t.y; dst[2] = t.z; dst[3] = t.w;
}
__device__ __forceinline__ float softplusf(float x) {
    return (x > 20.f) ? x : log1pf(__expf(x));
}

// load one K-step's A fragment + 4 B fragments (ISA 7.12.2 bf16 layouts)
__device__ __forceinline__ void load_step_frags(
    const bf16_t *__restrict__ A, const bf16_t *__restrict__ Wt,
    int K, int am, int m, int alo, int blo, int bn0, int lane, int k0,
    Frag16 &af, Frag16 bfm[4]) {
    const bf16_t *ap = A + (size_t)(am + m) * K + k0 + alo;
    load8bf(af.u,     ap);
    load8bf(af.u + 4, ap + 16);
#pragma unroll
    for (int j = 0; j < 4; ++j) {
        const bf16_t *bp = Wt + (size_t)(bn0 + j * 16 + (lane & 15)) * K + k0 + blo;
        load8bf(bfm[j].u,     bp);
        load8bf(bfm[j].u + 4, bp + 16);
    }
}

// ---------------- conversion kernels ----------------
__global__ __launch_bounds__(256) void f32_to_bf16_kernel(
    const float *__restrict__ in, bf16_t *__restrict__ out, int n) {
    int i = blockIdx.x * 256 + threadIdx.x;
    if (i < n) out[i] = (bf16_t)in[i];
}

// W is K x N row-major f32 -> Wt is N x K row-major bf16
__global__ __launch_bounds__(256) void transpose_bf16_kernel(
    const float *__restrict__ W, bf16_t *__restrict__ Wt, int K, int N) {
    int i = blockIdx.x * 256 + threadIdx.x;          // i = n*K + k
    int n = i / K, k = i - n * K;
    Wt[i] = (bf16_t)W[(size_t)k * N + n];
}

// ---------------- WMMA GEMM: C(MxN f32) = A(MxK bf16) @ Wt(NxK bf16)^T -------------
// EPI: 0 = plain store, 1 = softplus(x + bias), 2 = store + bf16 copy of cols < DTRANK
template <int EPI>
__global__ __launch_bounds__(256) void gemm_wmma_kernel(
    const bf16_t *__restrict__ A, const bf16_t *__restrict__ Wt,
    const float *__restrict__ bias, float *__restrict__ C,
    bf16_t *__restrict__ aux,
    int M, int N, int K) {
    const int NT = 4;                                 // 16x64 strip per wave
    int wave = blockIdx.x * (blockDim.x >> 5) + (threadIdx.x >> 5);
    int lane = threadIdx.x & 31;
    int ngrp = N / (16 * NT);
    int mtile = wave / ngrp;
    int ntile = wave - mtile * ngrp;
    if (mtile >= M / 16) return;                      // wave-uniform; EXEC all-1 for WMMA
    const int am  = mtile * 16;
    const int bn0 = ntile * 16 * NT;
    const int m   = lane & 15;
    const int alo = (lane < 16) ? 0 : 8;              // A: K {0..7,16..23} / {8..15,24..31}
    const int blo = (lane < 16) ? 0 : 8;              // B: K {0..7,16..23} / {8..15,24..31}

    v8f acc[NT];
    const v8f vz = {0.f, 0.f, 0.f, 0.f, 0.f, 0.f, 0.f, 0.f};
#pragma unroll
    for (int j = 0; j < NT; ++j) acc[j] = vz;

    // Software pipeline, 2x unrolled over K with STATICALLY-NAMED ping/pong
    // fragment sets (no runtime-indexed arrays -> everything stays in VGPRs).
    Frag16 a0, a1;
    Frag16 b0[NT], b1[NT];
    load_step_frags(A, Wt, K, am, m, alo, blo, bn0, lane, 0, a0, b0);
    for (int k0 = 0; k0 < K; k0 += 64) {
        if (k0 + 32 < K)                              // prefetch 2nd half-step
            load_step_frags(A, Wt, K, am, m, alo, blo, bn0, lane, k0 + 32, a1, b1);
        {
            const v16bf av = frag_as_v16bf(a0);
#pragma unroll
            for (int j = 0; j < NT; ++j)
                acc[j] = __builtin_amdgcn_wmma_f32_16x16x32_bf16(
                    false, av, false, frag_as_v16bf(b0[j]),
                    (short)0, acc[j], false, false);
        }
        if (k0 + 64 < K)                              // prefetch next iter's 1st half
            load_step_frags(A, Wt, K, am, m, alo, blo, bn0, lane, k0 + 64, a0, b0);
        if (k0 + 32 < K) {
            const v16bf av = frag_as_v16bf(a1);
#pragma unroll
            for (int j = 0; j < NT; ++j)
                acc[j] = __builtin_amdgcn_wmma_f32_16x16x32_bf16(
                    false, av, false, frag_as_v16bf(b1[j]),
                    (short)0, acc[j], false, false);
        }
    }
#pragma unroll
    for (int j = 0; j < NT; ++j) {
        const int n  = bn0 + j * 16 + (lane & 15);
        const float bv = bias ? bias[n] : 0.f;
#pragma unroll
        for (int v = 0; v < 8; ++v) {
            const int mm = am + v + ((lane < 16) ? 0 : 8);
            float r = acc[j][v] + bv;
            if (EPI == 1) r = softplusf(r);
            C[(size_t)mm * N + n] = r;
            if (EPI == 2 && n < DTRANK)               // fused dt-slice -> bf16
                aux[(size_t)mm * DTRANK + n] = (bf16_t)r;
        }
    }
}

// ---------------- depthwise conv (k=4, causal) + bias + silu ----------------
__global__ __launch_bounds__(256) void conv_silu_kernel(
    const float *__restrict__ xz, const float *__restrict__ cw,
    const float *__restrict__ cb, float *__restrict__ xs,
    bf16_t *__restrict__ xs_bf) {
    int idx = blockIdx.x * 256 + threadIdx.x;         // (b*L + l)*DINNER + d
    int d  = idx & (DINNER - 1);
    int bl = idx >> 10;
    int l  = bl & (LSEQ - 1);
    float acc = cb[d];
#pragma unroll
    for (int j = 0; j < DCONV; ++j) {
        int ll = l - (DCONV - 1) + j;
        if (ll >= 0)
            acc += cw[d * DCONV + j] * xz[(size_t)(bl - (l - ll)) * (2 * DINNER) + d];
    }
    float s = acc / (1.f + __expf(-acc));             // silu
    xs[idx]    = s;
    xs_bf[idx] = (bf16_t)s;
}

// ---------------- selective scan, one lane per (b,d,n) state ----------------
__global__ __launch_bounds__(256) void scan_kernel(
    const float *__restrict__ delta, const float *__restrict__ xs,
    const float *__restrict__ dbc,   const float *__restrict__ xz,
    const float *__restrict__ A_log, const float *__restrict__ Dsk,
    bf16_t *__restrict__ yg_bf) {
    int tid = blockIdx.x * 256 + threadIdx.x;
    int n   = tid & 15;
    int grp = tid >> 4;                               // (b,d)
    int d   = grp & (DINNER - 1);
    int b   = grp >> 10;
    const float A  = -__expf(A_log[d * DSTATE + n]);
    const float ds = Dsk[d];
    float h = 0.f;
    for (int t = 0; t < LSEQ; ++t) {
        const size_t row = (size_t)(b * LSEQ + t);
        if (t + 8 < LSEQ) {                           // pull t+8 toward the WGP
            const size_t prow = row + 8;              // (global_prefetch_b8)
            __builtin_prefetch(&delta[prow * DINNER + d], 0, 3);
            __builtin_prefetch(&xs[prow * DINNER + d], 0, 3);
            __builtin_prefetch(&dbc[prow * XPROJ_N + DTRANK + n], 0, 3);
        }
        const float dlt = delta[row * DINNER + d];
        const float xv  = xs[row * DINNER + d];
        const float Bn  = dbc[row * XPROJ_N + DTRANK + n];
        const float Cn  = dbc[row * XPROJ_N + DTRANK + DSTATE + n];
        h = __expf(dlt * A) * h + (dlt * xv) * Bn;    // dA*h + dBu
        float p = h * Cn;
        p += __shfl_xor(p, 8, 32);                    // reduce over 16 states
        p += __shfl_xor(p, 4, 32);
        p += __shfl_xor(p, 2, 32);
        p += __shfl_xor(p, 1, 32);
        if (n == 0) {
            const float z  = xz[row * (2 * DINNER) + DINNER + d];
            const float sg = z / (1.f + __expf(-z));  // silu(z)
            yg_bf[row * DINNER + d] = (bf16_t)((p + xv * ds) * sg);
        }
    }
}

// ---------------- residual add + LayerNorm + fused bf16 emit ----------------
__global__ __launch_bounds__(256) void add_ln_kernel(
    const float *__restrict__ mo, const float *__restrict__ res,
    const float *__restrict__ g,  const float *__restrict__ bta,
    float *__restrict__ out, bf16_t *__restrict__ out_bf) {
    __shared__ float s1[256], s2[256];
    const int row = blockIdx.x;
    const float *mr = mo  + (size_t)row * DMODEL;
    const float *rr = res + (size_t)row * DMODEL;
    const int tx = threadIdx.x;
    const float v0 = mr[tx]       + rr[tx];
    const float v1 = mr[tx + 256] + rr[tx + 256];
    s1[tx] = v0 + v1;
    s2[tx] = v0 * v0 + v1 * v1;
    __syncthreads();
    for (int o = 128; o > 0; o >>= 1) {
        if (tx < o) { s1[tx] += s1[tx + o]; s2[tx] += s2[tx + o]; }
        __syncthreads();
    }
    const float mean = s1[0] * (1.f / DMODEL);
    const float var  = s2[0] * (1.f / DMODEL) - mean * mean;
    const float inv  = rsqrtf(var + 1e-5f);
    const float r0 = (v0 - mean) * inv * g[tx]       + bta[tx];
    const float r1 = (v1 - mean) * inv * g[tx + 256] + bta[tx + 256];
    out[(size_t)row * DMODEL + tx]       = r0;
    out[(size_t)row * DMODEL + tx + 256] = r1;
    if (out_bf) {                                     // next layer's GEMM operand
        out_bf[(size_t)row * DMODEL + tx]       = (bf16_t)r0;
        out_bf[(size_t)row * DMODEL + tx + 256] = (bf16_t)r1;
    }
}

// ---------------- host-side orchestration ----------------
static inline size_t align256(size_t x) { return (x + 255) & ~(size_t)255; }

extern "C" void kernel_launch(void *const *d_in, const int *in_sizes, int n_in,
                              void *d_out, int out_size, void *d_ws, size_t ws_size,
                              hipStream_t stream) {
    (void)in_sizes; (void)n_in; (void)out_size; (void)ws_size;
    const float *x      = (const float *)d_in[0];
    const float *Wi     = (const float *)d_in[1];   // (L, 512, 2048)
    const float *cw     = (const float *)d_in[2];   // (L, 1024, 1, 4)
    const float *cb     = (const float *)d_in[3];   // (L, 1024)
    const float *Wx     = (const float *)d_in[4];   // (L, 1024, 64)
    const float *Wdt    = (const float *)d_in[5];   // (L, 32, 1024)
    const float *bdt    = (const float *)d_in[6];   // (L, 1024)
    const float *A_log  = (const float *)d_in[7];   // (L, 1024, 16)
    const float *Dsk    = (const float *)d_in[8];   // (L, 1024)
    const float *Wo     = (const float *)d_in[9];   // (L, 1024, 512)
    const float *ln_g   = (const float *)d_in[10];  // (L, 512)
    const float *ln_b   = (const float *)d_in[11];  // (L, 512)

    char *ws = (char *)d_ws;
    size_t off = 0;
    auto alloc = [&](size_t bytes) -> void * {
        void *p = ws + off; off = align256(off + bytes); return p;
    };
    bf16_t *wiT   = (bf16_t *)alloc((size_t)2048 * 512 * 2);
    bf16_t *wxT   = (bf16_t *)alloc((size_t)XPROJ_N * DINNER * 2);
    bf16_t *wdtT  = (bf16_t *)alloc((size_t)DINNER * DTRANK * 2);
    bf16_t *woT   = (bf16_t *)alloc((size_t)DMODEL * DINNER * 2);
    bf16_t *h_bf  = (bf16_t *)alloc((size_t)MROWS * DMODEL * 2);
    float  *xzbuf = (float  *)alloc((size_t)MROWS * 2 * DINNER * 4);
    float  *xsbuf = (float  *)alloc((size_t)MROWS * DINNER * 4);
    bf16_t *xs_bf = (bf16_t *)alloc((size_t)MROWS * DINNER * 2);
    float  *dbc   = (float  *)alloc((size_t)MROWS * XPROJ_N * 4);
    bf16_t *dt_bf = (bf16_t *)alloc((size_t)MROWS * DTRANK * 2);
    float  *delta = (float  *)alloc((size_t)MROWS * DINNER * 4);
    bf16_t *yg_bf = (bf16_t *)alloc((size_t)MROWS * DINNER * 2);
    float  *mo    = (float  *)alloc((size_t)MROWS * DMODEL * 4);
    float  *h_buf = (float  *)alloc((size_t)MROWS * DMODEL * 4);

    for (int l = 0; l < NLAYERS; ++l) {
        const float *hin  = (l == 0) ? x : h_buf;
        float *hout       = (l == NLAYERS - 1) ? (float *)d_out : h_buf;
        bf16_t *hout_bf   = (l == NLAYERS - 1) ? nullptr : h_bf;
        const float *Wi_l  = Wi    + (size_t)l * DMODEL * 2 * DINNER;
        const float *cw_l  = cw    + (size_t)l * DINNER * DCONV;
        const float *cb_l  = cb    + (size_t)l * DINNER;
        const float *Wx_l  = Wx    + (size_t)l * DINNER * XPROJ_N;
        const float *Wdt_l = Wdt   + (size_t)l * DTRANK * DINNER;
        const float *bdt_l = bdt   + (size_t)l * DINNER;
        const float *Al_l  = A_log + (size_t)l * DINNER * DSTATE;
        const float *Dsk_l = Dsk   + (size_t)l * DINNER;
        const float *Wo_l  = Wo    + (size_t)l * DINNER * DMODEL;
        const float *lg_l  = ln_g  + (size_t)l * DMODEL;
        const float *lb_l  = ln_b  + (size_t)l * DMODEL;

        // weight transpose + bf16
        transpose_bf16_kernel<<<(2048 * 512) / 256, 256, 0, stream>>>(Wi_l, wiT, DMODEL, 2 * DINNER);
        transpose_bf16_kernel<<<(XPROJ_N * DINNER) / 256, 256, 0, stream>>>(Wx_l, wxT, DINNER, XPROJ_N);
        transpose_bf16_kernel<<<(DINNER * DTRANK) / 256, 256, 0, stream>>>(Wdt_l, wdtT, DTRANK, DINNER);
        transpose_bf16_kernel<<<(DMODEL * DINNER) / 256, 256, 0, stream>>>(Wo_l, woT, DINNER, DMODEL);
        // layer-0 activation -> bf16 (later layers get it fused from add_ln)
        if (l == 0)
            f32_to_bf16_kernel<<<(MROWS * DMODEL) / 256, 256, 0, stream>>>(hin, h_bf, MROWS * DMODEL);

        // xz = h @ Wi            (M=4096, N=2048, K=512)
        gemm_wmma_kernel<0><<<(MROWS / 16) * (2 * DINNER / 64) / 8, 256, 0, stream>>>(
            h_bf, wiT, nullptr, xzbuf, nullptr, MROWS, 2 * DINNER, DMODEL);
        // conv + bias + silu
        conv_silu_kernel<<<(MROWS * DINNER) / 256, 256, 0, stream>>>(xzbuf, cw_l, cb_l, xsbuf, xs_bf);
        // dbc = xs @ Wx          (M=4096, N=64, K=1024), fused dt-slice -> bf16
        gemm_wmma_kernel<2><<<(MROWS / 16) * (XPROJ_N / 64) / 8, 256, 0, stream>>>(
            xs_bf, wxT, nullptr, dbc, dt_bf, MROWS, XPROJ_N, DINNER);
        // delta = softplus(dt @ Wdt + bdt)   (M=4096, N=1024, K=32)
        gemm_wmma_kernel<1><<<(MROWS / 16) * (DINNER / 64) / 8, 256, 0, stream>>>(
            dt_bf, wdtT, bdt_l, delta, nullptr, MROWS, DINNER, DTRANK);
        // selective scan (fused D-skip + silu(z) gating), emits bf16 operand
        scan_kernel<<<(BATCH * DINNER * DSTATE) / 256, 256, 0, stream>>>(
            delta, xsbuf, dbc, xzbuf, Al_l, Dsk_l, yg_bf);
        // mo = yg @ Wo           (M=4096, N=512, K=1024)
        gemm_wmma_kernel<0><<<(MROWS / 16) * (DMODEL / 64) / 8, 256, 0, stream>>>(
            yg_bf, woT, nullptr, mo, nullptr, MROWS, DMODEL, DINNER);
        // h = LN(mo + hin)  (+ fused bf16 emit for next layer's GEMM)
        add_ln_kernel<<<MROWS, 256, 0, stream>>>(mo, hin, lg_l, lb_l, hout, hout_bf);
    }
}
